// Net_29703993819346
// MI455X (gfx1250) — matrix operand order repaired and checked by reading, not compiled, not run on hardware
//
#include <hip/hip_runtime.h>
#include <hip/hip_bf16.h>

#define HID 128
#define RAWF 127

typedef __attribute__((ext_vector_type(16))) __bf16 v16bf;
typedef __attribute__((ext_vector_type(8)))  float  v8f;

static __device__ __forceinline__ __bf16 f2bf(float f) {
  union { float f; unsigned u; } v; v.f = f;
  unsigned r = v.u + 0x7FFFu + ((v.u >> 16) & 1u);   // round-to-nearest-even
  unsigned short s = (unsigned short)(r >> 16);
  __bf16 b; __builtin_memcpy(&b, &s, 2); return b;
}

__global__ void k_zero_f(float* __restrict__ p, int n) {
  int i = blockIdx.x * 256 + threadIdx.x;
  if (i < n) p[i] = 0.0f;
}

__global__ void k_deg(const int* __restrict__ dst, float* __restrict__ deg, int E) {
  int i = blockIdx.x * 256 + threadIdx.x;
  if (i < E) atomicAdd(deg + dst[i], 1.0f);
}

// h0 = concat([deg, feat]) : [N,128]
__global__ void k_h0(const float* __restrict__ feat, const float* __restrict__ deg,
                     float* __restrict__ h, int N) {
  int i = blockIdx.x * 256 + threadIdx.x;
  if (i >= N * HID) return;
  int n = i >> 7, c = i & 127;
  h[i] = (c == 0) ? deg[n] : feat[(size_t)n * RAWF + (c - 1)];
}

__global__ void k_w2bf(const float* __restrict__ W, __bf16* __restrict__ Wb, int n) {
  int i = blockIdx.x * 256 + threadIdx.x;
  if (i < n) Wb[i] = f2bf(W[i]);
}

// sum[dst] += h[src] ; one thread = (edge, 4-col chunk)
__global__ void k_scatter(const int* __restrict__ src, const int* __restrict__ dst,
                          const float* __restrict__ h, float* __restrict__ sum, int E) {
  int i = blockIdx.x * 256 + threadIdx.x;
  if (i >= E * 32) return;
  int e = i >> 5, c4 = (i & 31) << 2;
  int s = src[e], d = dst[e];
  const float4 v = *(const float4*)(h + (size_t)s * HID + c4);
  float* o = sum + (size_t)d * HID + c4;
  atomicAdd(o + 0, v.x); atomicAdd(o + 1, v.y);
  atomicAdd(o + 2, v.z); atomicAdd(o + 3, v.w);
}

// agg = deg>0 ? sum/deg : h  ->  bf16
__global__ void k_finalize(const float* __restrict__ sum, const float* __restrict__ h,
                           const float* __restrict__ deg, __bf16* __restrict__ hb, int N) {
  int i = blockIdx.x * 256 + threadIdx.x;
  if (i >= N * HID) return;
  int n = i >> 7;
  float d = deg[n];
  float v = (d > 0.0f) ? (sum[i] / d) : h[i];
  hb[i] = f2bf(v);
}

// out[N,128] = relu( hb[N,128] @ Wb[128,128]^T + bias )
// block = 256 threads = 8 waves; block -> 16-row node tile; wave -> 16-col output tile
__global__ void __launch_bounds__(256)
k_gemm_wmma(const __bf16* __restrict__ hb, const __bf16* __restrict__ Wb,
            const float* __restrict__ bias, float* __restrict__ out) {
  const int wave = threadIdx.x >> 5;
  const int lane = threadIdx.x & 31;
  const int half = lane >> 4;
  const int l16  = lane & 15;
  const int m0 = blockIdx.x * 16;     // node rows
  const int n0 = wave * 16;           // output cols

  v8f c = {};
#pragma unroll
  for (int kc = 0; kc < 4; ++kc) {
    v16bf a, b;
    // A fragment: row m = m0+l16 ; K base = kc*32 + 8*half
    const __bf16* arow = hb + (size_t)(m0 + l16) * HID + kc * 32 + half * 8;
#pragma unroll
    for (int e = 0; e < 16; ++e) {
      int j  = e >> 1;
      int kl = ((j & 3) << 1) + ((j >> 2) << 4) + (e & 1);
      a[e] = arow[kl];
    }
    // B fragment: B[k][n] = W[n][k]; col n = n0+l16 ; K base = kc*32 + 8*half
    const __bf16* brow = Wb + (size_t)(n0 + l16) * HID + kc * 32 + half * 8;
#pragma unroll
    for (int e = 0; e < 16; ++e) b[e] = brow[(e & 7) + ((e >> 3) << 4)];

    c = __builtin_amdgcn_wmma_f32_16x16x32_bf16(false, a, false, b, (short)0, c,
                                                false, false);
  }
  const float bv = bias[n0 + l16];
#pragma unroll
  for (int r = 0; r < 8; ++r) {
    float v = c[r] + bv;
    v = v > 0.0f ? v : 0.0f;
    out[(size_t)(m0 + r + half * 8) * HID + n0 + l16] = v;
  }
}

__global__ void k_counts(const int* __restrict__ gid, float* __restrict__ cnt, int N) {
  int i = blockIdx.x * 256 + threadIdx.x;
  if (i < N) atomicAdd(cnt + gid[i], 1.0f);
}

__global__ void k_pool_sum(const float* __restrict__ h, const int* __restrict__ gid,
                           float* __restrict__ gsum, int N) {
  int i = blockIdx.x * 256 + threadIdx.x;
  if (i >= N * HID) return;
  int n = i >> 7, c = i & 127;
  atomicAdd(gsum + (size_t)gid[n] * HID + c, h[i]);
}

__global__ void k_pool_div(float* __restrict__ gsum, const float* __restrict__ cnt, int G) {
  int i = blockIdx.x * 256 + threadIdx.x;
  if (i >= G * HID) return;
  float d = cnt[i >> 7];
  gsum[i] /= (d > 1.0f ? d : 1.0f);
}

// y[m,o] = b[o] + sum_k x[m,k] * W[o,k]   (block = 1 row, 128 threads)
__global__ void k_fc(const float* __restrict__ x, const float* __restrict__ W,
                     const float* __restrict__ b, float* __restrict__ y) {
  int m = blockIdx.x, o = threadIdx.x;
  const float* xr = x + (size_t)m * HID;
  const float* wr = W + (size_t)o * HID;
  float acc = b[o];
#pragma unroll 4
  for (int k = 0; k < HID; ++k) acc += xr[k] * wr[k];
  y[(size_t)m * HID + o] = acc;
}

__global__ void k_fc_out(const float* __restrict__ x, const float* __restrict__ W,
                         const float* __restrict__ b, float* __restrict__ y, int G) {
  int m = blockIdx.x * 256 + threadIdx.x;
  if (m >= G) return;
  const float* xr = x + (size_t)m * HID;
  float acc = b[0];
#pragma unroll 4
  for (int k = 0; k < HID; ++k) acc += xr[k] * W[k];
  y[m] = acc;
}

static inline size_t al256(size_t x) { return (x + 255) & ~(size_t)255; }

extern "C" void kernel_launch(void* const* d_in, const int* in_sizes, int n_in,
                              void* d_out, int out_size, void* d_ws, size_t ws_size,
                              hipStream_t stream) {
  const float* feat = (const float*)d_in[0];
  const int*   src  = (const int*)d_in[1];
  const int*   dst  = (const int*)d_in[2];
  const int*   gid  = (const int*)d_in[3];
  // d_in[4] = num_graphs (device scalar); use out_size instead (out is [G,1])
  const float* W[3] = {(const float*)d_in[5], (const float*)d_in[7], (const float*)d_in[9]};
  const float* B[3] = {(const float*)d_in[6], (const float*)d_in[8], (const float*)d_in[10]};
  const float* Wc1 = (const float*)d_in[11];
  const float* bc1 = (const float*)d_in[12];
  const float* Wc2 = (const float*)d_in[13];
  const float* bc2 = (const float*)d_in[14];

  const int N = in_sizes[3];          // 100000
  const int E = in_sizes[1];          // 1600000
  const int G = out_size;             // 100
  const int NH = N * HID;

  // workspace carve-up (~128.5 MB total)
  char* ws = (char*)d_ws;
  float*  deg  = (float*)ws;                 ws += al256((size_t)N * 4);
  float*  h    = (float*)ws;                 ws += al256((size_t)NH * 4);
  float*  sum  = (float*)ws;                 ws += al256((size_t)NH * 4);
  __bf16* hb   = (__bf16*)ws;                ws += al256((size_t)NH * 2);
  __bf16* Wb   = (__bf16*)ws;                ws += al256((size_t)HID * HID * 2);
  float*  gsum = (float*)ws;                 ws += al256((size_t)G * HID * 4);
  float*  cnt  = (float*)ws;                 ws += al256((size_t)G * 4);
  float*  hg1  = (float*)ws;                 ws += al256((size_t)G * HID * 4);
  float*  hg2  = (float*)ws;                 (void)ws_size;

  const int T = 256;
  #define GRID(n) dim3(((n) + T - 1) / T)

  // degrees
  k_zero_f<<<GRID(N), T, 0, stream>>>(deg, N);
  k_deg<<<GRID(E), T, 0, stream>>>(dst, deg, E);
  // h0 = [deg | feat]
  k_h0<<<GRID(NH), T, 0, stream>>>(feat, deg, h, N);

  // 3 GCN layers
  for (int l = 0; l < 3; ++l) {
    k_w2bf<<<GRID(HID * HID), T, 0, stream>>>(W[l], Wb, HID * HID);
    k_zero_f<<<GRID(NH), T, 0, stream>>>(sum, NH);
    k_scatter<<<GRID(E * 32), T, 0, stream>>>(src, dst, h, sum, E);
    k_finalize<<<GRID(NH), T, 0, stream>>>(sum, h, deg, hb, N);
    k_gemm_wmma<<<dim3(N / 16), T, 0, stream>>>(hb, Wb, B[l], h);
  }

  // per-graph mean pooling
  k_zero_f<<<GRID(G * HID), T, 0, stream>>>(gsum, G * HID);
  k_zero_f<<<GRID(G), T, 0, stream>>>(cnt, G);
  k_counts<<<GRID(N), T, 0, stream>>>(gid, cnt, N);
  k_pool_sum<<<GRID(NH), T, 0, stream>>>(h, gid, gsum, N);
  k_pool_div<<<GRID(G * HID), T, 0, stream>>>(gsum, cnt, G);

  // classifier: Wc1 applied twice (faithful to reference), then Wc2
  k_fc<<<dim3(G), dim3(HID), 0, stream>>>(gsum, Wc1, bc1, hg1);
  k_fc<<<dim3(G), dim3(HID), 0, stream>>>(hg1, Wc1, bc1, hg2);
  k_fc_out<<<GRID(G), T, 0, stream>>>(hg2, Wc2, bc2, (float*)d_out, G);
  #undef GRID
}
